// GCN_84318797955093
// MI455X (gfx1250) — compile-verified, matrix operand order checked by reading, not run on hardware
//
#include <hip/hip_runtime.h>

// ---------------------------------------------------------------------------
// CDNA5 (gfx1250) GCN pipeline, round 4.
// - v_wmma_f32_16x16x32_bf16, register double-buffered k-pipeline.
// - NT fixed at 4 tiles/wave: ~150 VGPRs -> no scratch spills (round 3's NT=8
//   spilled fragments to scratch inside the inner loop).
// - Wide outputs handled by column-groups (blockIdx.y), writing through a
//   real ldc into the full-width row.
// - Bt stored bf16 in fragment-permuted 32-k blocks: a lane's 16-element
//   fragment is ONE contiguous 32B (v16bf) load. Rows zero-padded to 32 | K.
// ---------------------------------------------------------------------------

typedef __attribute__((ext_vector_type(16))) __bf16 v16bf;
typedef __attribute__((ext_vector_type(8)))  float  v8f;

static __device__ __forceinline__ __bf16 bfc(float f) { return (__bf16)f; }

static __device__ __forceinline__ v16bf pack_a(float4 a0, float4 a1,
                                               float4 a2, float4 a3) {
  v16bf a;
  a[0]  = bfc(a0.x); a[1]  = bfc(a0.y); a[2]  = bfc(a0.z); a[3]  = bfc(a0.w);
  a[4]  = bfc(a1.x); a[5]  = bfc(a1.y); a[6]  = bfc(a1.z); a[7]  = bfc(a1.w);
  a[8]  = bfc(a2.x); a[9]  = bfc(a2.y); a[10] = bfc(a2.z); a[11] = bfc(a2.w);
  a[12] = bfc(a3.x); a[13] = bfc(a3.y); a[14] = bfc(a3.z); a[15] = bfc(a3.w);
  return a;
}

// Permuted position of k within its 32-block: 8-groups ordered [0,2,1,3] so
// lane sub-group s reads positions [16*s, 16*s+15] contiguously.
static __device__ __forceinline__ int kperm(int r) {   // r = k & 31
  int g = r >> 3;
  return ((g & 1) << 4) + ((g >> 1) << 3) + (r & 7);
}

// One wave: C[16 x NT*16] (+)= A[16 x krange] @ B[krange x NT*16].
// A f32 row-major (lda); Bt bf16 permuted-block rows (ldb multiple of 32,
// zero-padded past true K); C row stride ldc.
template <int NT, bool ATOMIC>
static __device__ __forceinline__ void
wave_gemm(const float* __restrict__ A, long lda,
          const __bf16* __restrict__ Bt, long ldb,
          float* __restrict__ C, long ldc, long m0, long kb, long ke) {
  const int lane = threadIdx.x & 31;
  const int sub  = lane >> 4;
  const int mr   = lane & 15;

  const float* __restrict__ Arow = A + (m0 + mr) * lda;
  const float* ap0 = Arow + kb + 8 * sub;
  const __bf16* bp[NT];
#pragma unroll
  for (int t = 0; t < NT; ++t)
    bp[t] = Bt + (size_t)(t * 16 + mr) * ldb + kb + 16 * sub;

  v8f acc[NT];
#pragma unroll
  for (int t = 0; t < NT; ++t) acc[t] = v8f{};

  const long nsteps = (ke - kb) / 32;

  float4 xa0, xa1, xa2, xa3; v16bf xb[NT];
  float4 ya0, ya1, ya2, ya3; v16bf yb[NT];

#define LOAD_SET(A0_, A1_, A2_, A3_, BF_, STEP_)                   \
  { const float* _ap = ap0 + (STEP_) * 32;                         \
    A0_ = *(const float4*)(_ap);                                   \
    A1_ = *(const float4*)(_ap + 4);                               \
    A2_ = *(const float4*)(_ap + 16);                              \
    A3_ = *(const float4*)(_ap + 20);                              \
    _Pragma("unroll")                                              \
    for (int _t = 0; _t < NT; ++_t)                                \
      BF_[_t] = *(const v16bf*)(bp[_t] + (STEP_) * 32); }

#define WMMA_SET(A0_, A1_, A2_, A3_, BF_)                          \
  { v16bf _a = pack_a(A0_, A1_, A2_, A3_);                         \
    _Pragma("unroll")                                              \
    for (int _t = 0; _t < NT; ++_t)                                \
      acc[_t] = __builtin_amdgcn_wmma_f32_16x16x32_bf16(           \
          false, _a, false, BF_[_t], (short)0, acc[_t], false, false); }

  long s = 0;
  if (nsteps > 0) LOAD_SET(xa0, xa1, xa2, xa3, xb, 0);
  for (; s + 2 <= nsteps; s += 2) {
    LOAD_SET(ya0, ya1, ya2, ya3, yb, s + 1);
    WMMA_SET(xa0, xa1, xa2, xa3, xb);
    if (s + 2 < nsteps) LOAD_SET(xa0, xa1, xa2, xa3, xb, s + 2);
    WMMA_SET(ya0, ya1, ya2, ya3, yb);
  }
  if (s < nsteps) WMMA_SET(xa0, xa1, xa2, xa3, xb);
#undef LOAD_SET
#undef WMMA_SET

  // Tail (K % 32 != 0, final split-K chunk only); B zero-padded -> only A
  // loads need guards.
  const long kf = kb + nsteps * 32;
  if (kf < ke) {
    v16bf a;
#pragma unroll
    for (int i = 0; i < 16; ++i) {
      long k = kf + ((i >= 8) ? 16 : 0) + (i & 7) + 8 * sub;
      a[i] = bfc((k < ke) ? Arow[k] : 0.0f);
    }
#pragma unroll
    for (int t = 0; t < NT; ++t) {
      v16bf b = *(const v16bf*)(bp[t] + (kf - kb));
      acc[t] = __builtin_amdgcn_wmma_f32_16x16x32_bf16(
          false, a, false, b, (short)0, acc[t], false, false);
    }
  }

  // Store C tiles: vgpr r -> row (r + 8*sub), col = t*16 + (lane & 15)
#pragma unroll
  for (int t = 0; t < NT; ++t) {
#pragma unroll
    for (int r = 0; r < 8; ++r) {
      float* p = C + (m0 + r + 8 * sub) * ldc + t * 16 + mr;
      if (ATOMIC) atomicAdd(p, acc[t][r]);
      else        *p = acc[t][r];
    }
  }
}

// blockIdx.x: 16-row stripe; blockIdx.y: column group of NT*16 columns.
template <int NT>
__global__ __launch_bounds__(32)
void k_gemm(const float* __restrict__ A, long lda,
            const __bf16* __restrict__ Bt, long ldb,
            float* __restrict__ C, long ldc, int K) {
  long n0 = (long)blockIdx.y * (NT * 16);
  wave_gemm<NT, false>(A, lda, Bt + n0 * ldb, ldb, C + n0, ldc,
                       (long)blockIdx.x * 16, 0, K);
}

// blockIdx.x: 16-row stripe; blockIdx.y: split-K chunk. Single column group.
template <int NT>
__global__ __launch_bounds__(32)
void k_gemm_splitk(const float* __restrict__ A, long lda,
                   const __bf16* __restrict__ Bt, long ldb,
                   float* __restrict__ C, long ldc, int K, int kChunk) {
  long kb = (long)blockIdx.y * kChunk;
  long ke = kb + kChunk;
  if (ke > (long)K) ke = (long)K;
  if (kb >= ke) return;
  wave_gemm<NT, true>(A, lda, Bt, ldb, C, ldc, (long)blockIdx.x * 16, kb, ke);
}

// ---------------------------------------------------------------------------
// Scalar / aggregation kernels
// ---------------------------------------------------------------------------

__global__ void k_init(float* __restrict__ deg, int n,
                       float* __restrict__ out, int outn) {
  int i = blockIdx.x * blockDim.x + threadIdx.x;
  if (i < n)    deg[i] = 1.0f;
  if (i < outn) out[i] = 0.0f;
}

__global__ void k_deg(const long long* __restrict__ ei, long E,
                      float* __restrict__ deg) {
  long e = (long)blockIdx.x * blockDim.x + threadIdx.x;
  if (e < E) atomicAdd(&deg[(int)ei[E + e]], 1.0f);
}

__global__ void k_rsqrt(float* __restrict__ deg, int n) {
  int i = blockIdx.x * blockDim.x + threadIdx.x;
  if (i < n) deg[i] = rsqrtf(fmaxf(deg[i], 1.0f));
}

// Weights: B[K,N] row-major -> Bt[N][K] bf16, permuted-block k layout.
__global__ void k_wt_bf(const float* __restrict__ B, __bf16* __restrict__ Bt,
                        int K, int Ncols) {
  int idx = blockIdx.x * blockDim.x + threadIdx.x;
  if (idx >= K * Ncols) return;
  int k = idx / Ncols, n = idx - k * Ncols;
  int pos = (k & ~31) + kperm(k & 31);
  Bt[(size_t)n * K + pos] = bfc(B[idx]);
}

// Layer-1 aggregation: row-major [N, F]
__global__ void k_selfloop(const float* __restrict__ h, const float* __restrict__ dinv,
                           float* __restrict__ out, long total, int F) {
  long idx = (long)blockIdx.x * blockDim.x + threadIdx.x;
  if (idx >= total) return;
  float d = dinv[idx / F];
  out[idx] = h[idx] * d * d;
}

__global__ void k_edge_agg(const float* __restrict__ h,
                           const long long* __restrict__ ei, long E,
                           const float* __restrict__ dinv,
                           float* __restrict__ out, int F) {
  long idx = (long)blockIdx.x * blockDim.x + threadIdx.x;
  if (idx >= E * (long)F) return;
  long e = idx / F;
  int  f = (int)(idx - e * F);
  int  s = (int)ei[e], d = (int)ei[E + e];
  atomicAdd(&out[(long)d * F + f], h[(long)s * F + f] * dinv[s] * dinv[d]);
}

__global__ void k_bias_relu(float* __restrict__ out, const float* __restrict__ b,
                            long total, int F) {
  long idx = (long)blockIdx.x * blockDim.x + threadIdx.x;
  if (idx >= total) return;
  out[idx] = fmaxf(out[idx] + b[(int)(idx % F)], 0.0f);
}

// Layer-2 aggregation directly into TRANSPOSED f32 layout [F][N]
__global__ void k_selfloop_t(const float* __restrict__ h, const float* __restrict__ dinv,
                             float* __restrict__ outT, long total, int F, long N) {
  long idx = (long)blockIdx.x * blockDim.x + threadIdx.x;
  if (idx >= total) return;
  long i = idx / F;
  int  f = (int)(idx - i * F);
  float d = dinv[i];
  outT[(long)f * N + i] = h[idx] * d * d;
}

__global__ void k_edge_agg_t(const float* __restrict__ h,
                             const long long* __restrict__ ei, long E,
                             const float* __restrict__ dinv,
                             float* __restrict__ outT, int F, long N) {
  long idx = (long)blockIdx.x * blockDim.x + threadIdx.x;
  if (idx >= E * (long)F) return;
  long e = idx / F;
  int  f = (int)(idx - e * F);
  int  s = (int)ei[e], d = (int)ei[E + e];
  atomicAdd(&outT[(long)f * N + d], h[(long)s * F + f] * dinv[s] * dinv[d]);
}

// [F][N] f32 + bias -> [F][Kp] bf16 permuted-block, zero-padded to Kp.
__global__ void k_bias_cvt_perm(const float* __restrict__ inT,
                                const float* __restrict__ b,
                                __bf16* __restrict__ outBf,
                                long N, long Kp, int F) {
  long idx = (long)blockIdx.x * blockDim.x + threadIdx.x;   // over F*Kp
  if (idx >= (long)F * Kp) return;
  long f = idx / Kp;
  long k = idx - f * Kp;
  float v = (k < N) ? inT[f * N + k] + b[(int)f] : 0.0f;
  long pos = (k & ~31L) + kperm((int)(k & 31));
  outBf[f * Kp + pos] = bfc(v);
}

// ---------------------------------------------------------------------------
// Host-side launch
// ---------------------------------------------------------------------------

extern "C" void kernel_launch(void* const* d_in, const int* in_sizes, int n_in,
                              void* d_out, int out_size, void* d_ws, size_t ws_size,
                              hipStream_t stream) {
  const float*     data = (const float*)d_in[0];      // [B, N]
  const float*     x    = (const float*)d_in[1];      // [N, GENE]
  const long long* ei   = (const long long*)d_in[2];  // [2, E] int64
  const float*     W1   = (const float*)d_in[3];      // [GENE, HID]
  const float*     b1   = (const float*)d_in[4];      // [HID]
  const float*     W2   = (const float*)d_in[5];      // [HID, FEAT]
  const float*     b2   = (const float*)d_in[6];      // [FEAT]
  float*           out  = (float*)d_out;              // [B, FEAT]

  const int GENE = 256;
  const int HID  = in_sizes[4];            // 128
  const int FEAT = in_sizes[6];            // 64
  const int N    = in_sizes[1] / GENE;     // 50000
  const long E   = (long)in_sizes[2] / 2;  // 800000
  const int B    = in_sizes[0] / N;        // 1024
  const long Kp  = ((long)N + 31) & ~31L;  // 50016, B-pad for GEMM3

  // Workspace (floats): dinv[N] | bufA[N*HID] | bufB[N*HID] | weights(bf16)
  float* ws    = (float*)d_ws;
  float* dinv  = ws;
  float* h1    = ws + N;                         // [N, HID] f32
  float* bufB  = ws + N + (long)N * HID;         // N*HID floats
  float* h1agg = bufB;                           // [N, HID] f32
  float* h2    = h1;                             // [N, FEAT] f32 (h1 dead)
  float* aggT  = bufB;                           // [FEAT, N] f32 (h1agg dead)
  __bf16* aggTbf = (__bf16*)(bufB + (long)N * FEAT);     // [FEAT, Kp] bf16
  __bf16* W1t  = (__bf16*)(ws + N + 2 * (long)N * HID);  // [HID, GENE]
  __bf16* W2t  = W1t + (size_t)HID * GENE;               // [FEAT, HID]

  const int TB = 256;

  // 0) init
  {
    int mx = (N > out_size) ? N : out_size;
    k_init<<<(mx + TB - 1) / TB, TB, 0, stream>>>(dinv, N, out, out_size);
  }
  // 1) degrees + rsqrt
  k_deg<<<(int)((E + TB - 1) / TB), TB, 0, stream>>>(ei, E, dinv);
  k_rsqrt<<<(N + TB - 1) / TB, TB, 0, stream>>>(dinv, N);

  // 2) weights -> transposed, bf16, fragment-permuted
  k_wt_bf<<<(GENE * HID + TB - 1) / TB, TB, 0, stream>>>(W1, W1t, GENE, HID);
  k_wt_bf<<<(HID * FEAT + TB - 1) / TB, TB, 0, stream>>>(W2, W2t, HID, FEAT);

  // 3) GEMM1: h1 = x @ W1  (two 64-col groups; x re-read hits L2)
  k_gemm<4><<<dim3(N / 16, HID / 64), 32, 0, stream>>>(x, GENE, W1t, GENE, h1, HID, GENE);

  // 4) aggregate layer 1 + bias + ReLU
  {
    long tot = (long)N * HID;
    k_selfloop<<<(int)((tot + TB - 1) / TB), TB, 0, stream>>>(h1, dinv, h1agg, tot, HID);
    long etot = E * HID;
    k_edge_agg<<<(int)((etot + TB - 1) / TB), TB, 0, stream>>>(h1, ei, E, dinv, h1agg, HID);
    k_bias_relu<<<(int)((tot + TB - 1) / TB), TB, 0, stream>>>(h1agg, b1, tot, HID);
  }

  // 5) GEMM2: h2 = h1agg @ W2  (single 64-col group)
  k_gemm<4><<<dim3(N / 16, 1), 32, 0, stream>>>(h1agg, HID, W2t, HID, h2, FEAT, HID);

  // 6) aggregate layer 2 into [FEAT][N] f32, then bias + cvt -> permuted bf16
  {
    long tot = (long)N * FEAT;
    k_selfloop_t<<<(int)((tot + TB - 1) / TB), TB, 0, stream>>>(h2, dinv, aggT, tot, FEAT, N);
    long etot = E * FEAT;
    k_edge_agg_t<<<(int)((etot + TB - 1) / TB), TB, 0, stream>>>(h2, ei, E, dinv, aggT, FEAT, N);
    long tot2 = (long)FEAT * Kp;
    k_bias_cvt_perm<<<(int)((tot2 + TB - 1) / TB), TB, 0, stream>>>(aggT, b2, aggTbf, N, Kp, FEAT);
  }

  // 7) GEMM3 (HBM-bound: 205 MB of `data`, read exactly once):
  //    out = data @ h2agg; split-K (1024 -> 49 chunks, ~3.1K waves) + atomics.
  {
    const int kChunk = 1024;
    const int kSplit = (N + kChunk - 1) / kChunk;   // 49
    k_gemm_splitk<4><<<dim3(B / 16, kSplit), 32, 0, stream>>>(
        data, N, aggTbf, Kp, out, FEAT, N, kChunk);
  }
}
